// RnnSenderGS_31980326486496
// MI455X (gfx1250) — compile-verified, matrix operand order checked by reading, not run on hardware
//
#include <hip/hip_runtime.h>
#include <hip/hip_bf16.h>

typedef __bf16 bf16_t;
typedef __attribute__((ext_vector_type(16))) __bf16 v16bf;
typedef __attribute__((ext_vector_type(8)))  __bf16 v8bf;
typedef __attribute__((ext_vector_type(8)))  float  v8f;

#define MIN_REAL (-3.4028234663852886e+38f)

// ---- weight layout in workspace (bf16 elements) ----
#define OFF_WAGENT 0               // [512][512]
#define OFF_WIH    262144          // [1536][256]
#define OFF_WHH    655360          // [1536][512]
#define OFF_WOUT   1441792         // [512][512]
#define OFF_WEMB   1703936         // [256][512]
#define W_TOTAL    1835008

// Convert all five weight matrices to bf16 into the workspace.
__global__ void convert_weights_kernel(const float* __restrict__ Wa,
                                       const float* __restrict__ Wih,
                                       const float* __restrict__ Whh,
                                       const float* __restrict__ Wout,
                                       const float* __restrict__ Wemb,
                                       bf16_t* __restrict__ wbf) {
    for (int i = blockIdx.x * blockDim.x + threadIdx.x; i < W_TOTAL;
         i += gridDim.x * blockDim.x) {
        float v;
        if      (i < OFF_WIH)  v = Wa[i - OFF_WAGENT];
        else if (i < OFF_WHH)  v = Wih[i - OFF_WIH];
        else if (i < OFF_WOUT) v = Whh[i - OFF_WHH];
        else if (i < OFF_WEMB) v = Wout[i - OFF_WOUT];
        else                   v = Wemb[i - OFF_WEMB];
        wbf[i] = (bf16_t)v;
    }
}

// Load a 16x32 bf16 fragment (A or B operand) from a row-major [16+][ld]
// array. Lane l holds row (l&15); K chunks follow the CDNA5 16-bit A-matrix
// layout: VGPR0-3 = K[half*8 .. half*8+7], VGPR4-7 = K[16+half*8 .. 16+half*8+7].
__device__ __forceinline__ v16bf load_frag16(const bf16_t* __restrict__ base,
                                             int ld, int k0, int lane) {
    const int r = lane & 15, half = lane >> 4;
    const bf16_t* p0 = base + r * ld + k0 + half * 8;
    v8bf lo = *(const v8bf*)p0;
    v8bf hi = *(const v8bf*)(p0 + 16);
    return __builtin_shufflevector(lo, hi, 0, 1, 2, 3, 4, 5, 6, 7,
                                   8, 9, 10, 11, 12, 13, 14, 15);
}

// Dual-tile accumulate: C0 += A*W0^T, C1 += A*W1^T sharing one A-fragment.
// Two independent accumulator chains keep the XDL pipe busy and halve LDS
// A-fragment traffic.
__device__ __forceinline__ void gemm_acc2(v8f& c0, v8f& c1,
                                          const bf16_t* __restrict__ A, int lda,
                                          const bf16_t* __restrict__ W0,
                                          const bf16_t* __restrict__ W1, int K,
                                          int lane) {
    for (int k0 = 0; k0 < K; k0 += 32) {
        v16bf a  = load_frag16(A, lda, k0, lane);
        v16bf b0 = load_frag16(W0, K, k0, lane);
        v16bf b1 = load_frag16(W1, K, k0, lane);
        c0 = __builtin_amdgcn_wmma_f32_16x16x32_bf16(false, a, false, b0,
                                                     (short)0, c0, false, false);
        c1 = __builtin_amdgcn_wmma_f32_16x16x32_bf16(false, a, false, b1,
                                                     (short)0, c1, false, false);
    }
}

// D layout: VGPR i, lanes 0-15 -> M=i, N=lane; lanes 16-31 -> M=i+8, N=lane-16.
__device__ __forceinline__ void store_tile_f32(float* __restrict__ out, int ld,
                                               int n0, int lane, const v8f& c) {
    const int col = lane & 15, rs = lane >> 4;
#pragma unroll
    for (int i = 0; i < 8; ++i) out[(i + 8 * rs) * ld + n0 + col] = c[i];
}

__device__ __forceinline__ v8f splat8(float v) {
    v8f c;
#pragma unroll
    for (int i = 0; i < 8; ++i) c[i] = v;
    return c;
}

__global__ __launch_bounds__(256) void rnn_sender_kernel(
    const float* __restrict__ x, const float* __restrict__ noise_u,
    const float* __restrict__ b_agent, const float* __restrict__ b_ih,
    const float* __restrict__ b_hh, const float* __restrict__ b_out,
    const float* __restrict__ b_emb, const float* __restrict__ sos,
    const bf16_t* __restrict__ wbf, float* __restrict__ out_seq,
    float* __restrict__ out_ent) {
    // LDS: 152 KB static -> 2 workgroups per WGP (<=320 KB/WGP on CDNA5).
    __shared__ __align__(16) unsigned char smem[155648];
    float*  g_rz  = (float*)smem;                   // [16][1024] fused r,z gates
    float*  buf_a = (float*)(smem + 65536);         // [16][512] i_n / logits
    float*  buf_b = (float*)(smem + 98304);         // [16][512] h_n / h0 tmp
    bf16_t* h_bf  = (bf16_t*)(smem + 131072);       // [16][512]
    bf16_t* e_bf  = (bf16_t*)(smem + 147456);       // [16][256]
    bf16_t* s_bf  = (bf16_t*)smem;                  // alias g_rz: [16][512] sample / x

    const int tid  = threadIdx.x;
    const int lane = tid & 31;
    const int wave = tid >> 5;
    const int col  = lane & 15;
    const int rs   = lane >> 4;
    const int row0 = blockIdx.x * 16;

    const bf16_t* Wagent = wbf + OFF_WAGENT;
    const bf16_t* Wih    = wbf + OFF_WIH;
    const bf16_t* Whh    = wbf + OFF_WHH;
    const bf16_t* Wout   = wbf + OFF_WOUT;
    const bf16_t* Wemb   = wbf + OFF_WEMB;

    // ---- init: x tile -> bf16 (in s_bf), e0 = sos ----
    for (int i = tid; i < 16 * 512; i += 256) {
        int r = i >> 9, c = i & 511;
        s_bf[i] = (bf16_t)x[(size_t)(row0 + r) * 512 + c];
    }
    for (int i = tid; i < 16 * 256; i += 256)
        e_bf[i] = (bf16_t)sos[i & 255];
    __syncthreads();

    // ---- h0 = x @ Wagent^T + b_agent  (N=512: 16 tile-pairs) ----
    for (int p = wave; p < 16; p += 8) {
        int n0 = p * 32, n1 = n0 + 16;
        v8f c0 = splat8(b_agent[n0 + col]);
        v8f c1 = splat8(b_agent[n1 + col]);
        gemm_acc2(c0, c1, s_bf, 512, Wagent + (size_t)n0 * 512,
                  Wagent + (size_t)n1 * 512, 512, lane);
#pragma unroll
        for (int i = 0; i < 8; ++i) {
            int r = i + 8 * rs;
            buf_b[r * 512 + n0 + col] = c0[i];
            h_bf[r * 512 + n0 + col]  = (bf16_t)c0[i];
            buf_b[r * 512 + n1 + col] = c1[i];
            h_bf[r * 512 + n1 + col]  = (bf16_t)c1[i];
        }
    }
    __syncthreads();

    float h_reg[32];
#pragma unroll
    for (int i = 0; i < 32; ++i) h_reg[i] = buf_b[tid + 256 * i];

    // ---- recurrence ----
    for (int step = 0; step < 32; ++step) {
        __syncthreads();

        // g_rz = (e@Wih^T + b_ih + h@Whh^T + b_hh)[:, 0:1024]  (N=1024: 32 pairs)
        for (int p = wave; p < 32; p += 8) {
            int n0 = p * 32, n1 = n0 + 16;
            v8f c0 = splat8(b_ih[n0 + col] + b_hh[n0 + col]);
            v8f c1 = splat8(b_ih[n1 + col] + b_hh[n1 + col]);
            gemm_acc2(c0, c1, e_bf, 256, Wih + (size_t)n0 * 256,
                      Wih + (size_t)n1 * 256, 256, lane);
            gemm_acc2(c0, c1, h_bf, 512, Whh + (size_t)n0 * 512,
                      Whh + (size_t)n1 * 512, 512, lane);
            store_tile_f32(g_rz, 1024, n0, lane, c0);
            store_tile_f32(g_rz, 1024, n1, lane, c1);
        }
        // i_n = e @ Wih[1024:1536]^T + b_ih[1024:]  (N=512: 16 pairs)
        for (int p = wave; p < 16; p += 8) {
            int n0 = p * 32, n1 = n0 + 16;
            v8f c0 = splat8(b_ih[1024 + n0 + col]);
            v8f c1 = splat8(b_ih[1024 + n1 + col]);
            gemm_acc2(c0, c1, e_bf, 256, Wih + (size_t)(1024 + n0) * 256,
                      Wih + (size_t)(1024 + n1) * 256, 256, lane);
            store_tile_f32(buf_a, 512, n0, lane, c0);
            store_tile_f32(buf_a, 512, n1, lane, c1);
        }
        // h_n = h @ Whh[1024:1536]^T + b_hh[1024:]
        for (int p = wave; p < 16; p += 8) {
            int n0 = p * 32, n1 = n0 + 16;
            v8f c0 = splat8(b_hh[1024 + n0 + col]);
            v8f c1 = splat8(b_hh[1024 + n1 + col]);
            gemm_acc2(c0, c1, h_bf, 512, Whh + (size_t)(1024 + n0) * 512,
                      Whh + (size_t)(1024 + n1) * 512, 512, lane);
            store_tile_f32(buf_b, 512, n0, lane, c0);
            store_tile_f32(buf_b, 512, n1, lane, c1);
        }
        __syncthreads();

        // GRU elementwise (h in registers)
#pragma unroll
        for (int i = 0; i < 32; ++i) {
            int idx = tid + 256 * i;
            int r = idx >> 9, c = idx & 511;
            float rr = 1.f / (1.f + __expf(-g_rz[r * 1024 + c]));
            float zz = 1.f / (1.f + __expf(-g_rz[r * 1024 + 512 + c]));
            float nn = tanhf(buf_a[idx] + rr * buf_b[idx]);
            float hv = (1.f - zz) * nn + zz * h_reg[i];
            h_reg[i] = hv;
            h_bf[idx] = (bf16_t)hv;
        }
        __syncthreads();

        // logits = h @ Wout^T + b_out  -> buf_a  (N=512: 16 pairs)
        for (int p = wave; p < 16; p += 8) {
            int n0 = p * 32, n1 = n0 + 16;
            v8f c0 = splat8(b_out[n0 + col]);
            v8f c1 = splat8(b_out[n1 + col]);
            gemm_acc2(c0, c1, h_bf, 512, Wout + (size_t)n0 * 512,
                      Wout + (size_t)n1 * 512, 512, lane);
            store_tile_f32(buf_a, 512, n0, lane, c0);
            store_tile_f32(buf_a, 512, n1, lane, c1);
        }
        __syncthreads();

        // Gumbel softmax + entropy: 16 lanes per row (half-wave reductions).
        {
            const int r = tid >> 4, t = tid & 15;
            const float* nrow =
                noise_u + ((size_t)step * 4096 + row0 + r) * 512;
            float y[32];
            float mx = -3.4e38f;
#pragma unroll
            for (int j = 0; j < 32; ++j) {
                int c = t + j * 16;
                float u = nrow[c];
                float g = -__logf(-__logf(u));
                float v = buf_a[r * 512 + c] + g;  // TEMP == 1
                y[j] = v;
                mx = fmaxf(mx, v);
            }
#pragma unroll
            for (int m = 1; m < 16; m <<= 1) mx = fmaxf(mx, __shfl_xor(mx, m, 32));
            float s = 0.f;
#pragma unroll
            for (int j = 0; j < 32; ++j) { y[j] = __expf(y[j] - mx); s += y[j]; }
#pragma unroll
            for (int m = 1; m < 16; m <<= 1) s += __shfl_xor(s, m, 32);
            float inv = 1.f / s;
            float ent = 0.f;
            float* orow = out_seq + ((size_t)(row0 + r) * 32 + step) * 512;
#pragma unroll
            for (int j = 0; j < 32; ++j) {
                int c = t + j * 16;
                float p = y[j] * inv;
                orow[c] = p;
                s_bf[r * 512 + c] = (bf16_t)p;
                float lp = fmaxf(__log2f(p), MIN_REAL);
                ent -= p * lp;
            }
#pragma unroll
            for (int m = 1; m < 16; m <<= 1) ent += __shfl_xor(ent, m, 32);
            if (t == 0) out_ent[(size_t)(row0 + r) * 32 + step] = ent;
        }
        __syncthreads();

        // e_next = sample @ Wemb^T + b_emb  (N=256: 8 pairs) -> e_bf (bf16)
        {
            int p = wave;  // exactly 8 pairs, one per wave
            int n0 = p * 32, n1 = n0 + 16;
            v8f c0 = splat8(b_emb[n0 + col]);
            v8f c1 = splat8(b_emb[n1 + col]);
            gemm_acc2(c0, c1, s_bf, 512, Wemb + (size_t)n0 * 512,
                      Wemb + (size_t)n1 * 512, 512, lane);
#pragma unroll
            for (int i = 0; i < 8; ++i) {
                int r = i + 8 * rs;
                e_bf[r * 256 + n0 + col] = (bf16_t)c0[i];
                e_bf[r * 256 + n1 + col] = (bf16_t)c1[i];
            }
        }
    }
}

extern "C" void kernel_launch(void* const* d_in, const int* in_sizes, int n_in,
                              void* d_out, int out_size, void* d_ws, size_t ws_size,
                              hipStream_t stream) {
    (void)in_sizes; (void)n_in; (void)out_size; (void)ws_size;
    const float* x       = (const float*)d_in[0];
    const float* noise_u = (const float*)d_in[1];
    const float* W_agent = (const float*)d_in[2];
    const float* b_agent = (const float*)d_in[3];
    const float* W_ih    = (const float*)d_in[4];
    const float* W_hh    = (const float*)d_in[5];
    const float* b_ih    = (const float*)d_in[6];
    const float* b_hh    = (const float*)d_in[7];
    const float* W_out   = (const float*)d_in[8];
    const float* b_out   = (const float*)d_in[9];
    const float* W_emb   = (const float*)d_in[10];
    const float* b_emb   = (const float*)d_in[11];
    const float* sos     = (const float*)d_in[12];

    bf16_t* wbf = (bf16_t*)d_ws;
    float* out_seq = (float*)d_out;
    float* out_ent = out_seq + (size_t)4096 * 32 * 512;

    convert_weights_kernel<<<896, 256, 0, stream>>>(W_agent, W_ih, W_hh,
                                                    W_out, W_emb, wbf);
    rnn_sender_kernel<<<256, 256, 0, stream>>>(x, noise_u, b_agent, b_ih, b_hh,
                                               b_out, b_emb, sos, wbf,
                                               out_seq, out_ent);
}
